// Block_14499809591408
// MI455X (gfx1250) — matrix operand order repaired and checked by reading, not compiled, not run on hardware
//
#include <hip/hip_runtime.h>
#include <hip/hip_bf16.h>

#define BATCHN 2
#define SEQN   2048
#define HEADSN 8
#define HDIMN  64
#define HID    512
#define ROWSN  (BATCHN * SEQN)   // 4096
#define QKVN   3584              // q,k,v,p columns of GEMM1
#define A2C    2560              // concat(o, gelu(p))

typedef __attribute__((ext_vector_type(16))) __bf16 v16bf;
typedef __attribute__((ext_vector_type(8))) float v8f;

struct Frag32 { uint4 a, b; };  // 32 bytes = one WMMA 16-bit operand

static __device__ __forceinline__ unsigned short f32_to_bf16u(float f) {
  unsigned int u = __builtin_bit_cast(unsigned int, f);
  u += 0x7FFFu + ((u >> 16) & 1u);
  return (unsigned short)(u >> 16);
}

// Async DMA: global (16B per lane) -> LDS, tracked by ASYNCcnt (no VGPR round-trip).
static __device__ __forceinline__ void async_b128(unsigned lds_byte_addr, const void* gaddr) {
  asm volatile("global_load_async_to_lds_b128 %0, %1, off"
               :: "v"(lds_byte_addr), "v"(gaddr) : "memory");
}
static __device__ __forceinline__ void wait_async0() {
  asm volatile("s_wait_asynccnt 0x0" ::: "memory");
}
static __device__ __forceinline__ unsigned lds_addr32(const void* p) {
  return (unsigned)(uintptr_t)p;  // low 32 bits of flat shared ptr == LDS byte offset
}

// A-operand fragment: per-lane two contiguous 8-element runs at K=8h and K=16+8h.
static __device__ __forceinline__ v16bf load_afrag(const unsigned short* base, int h) {
  Frag32 fr;
  fr.a = *(const uint4*)(base + 8 * h);
  fr.b = *(const uint4*)(base + 16 + 8 * h);
  return __builtin_bit_cast(v16bf, fr);
}
// B-operand fragment: per-lane two contiguous 8-element runs at K=8h and K=16+8h.
static __device__ __forceinline__ v16bf load_bfrag(const unsigned short* base, int h) {
  Frag32 fr;
  fr.a = *(const uint4*)(base + 8 * h);
  fr.b = *(const uint4*)(base + 16 + 8 * h);
  return __builtin_bit_cast(v16bf, fr);
}

// ---------------- LayerNorm: one block per (b,l) row ----------------
__global__ __launch_bounds__(256) void ln_kernel(const float* __restrict__ x,
                                                 const float* __restrict__ gamma,
                                                 const float* __restrict__ beta,
                                                 float* __restrict__ xn) {
  int row = blockIdx.x;
  int t = threadIdx.x;
  int lane = t & 31, wid = t >> 5;
  const float* xr = x + (size_t)row * HID;
  float v0 = xr[t], v1 = xr[t + 256];
  float s = v0 + v1;
#pragma unroll
  for (int m = 16; m > 0; m >>= 1) s += __shfl_xor(s, m);
  __shared__ float red[8];
  if (lane == 0) red[wid] = s;
  __syncthreads();
  float tot = 0.f;
#pragma unroll
  for (int i = 0; i < 8; i++) tot += red[i];
  float mu = tot * (1.0f / HID);
  __syncthreads();
  float d0 = v0 - mu, d1 = v1 - mu;
  float s2 = d0 * d0 + d1 * d1;
#pragma unroll
  for (int m = 16; m > 0; m >>= 1) s2 += __shfl_xor(s2, m);
  if (lane == 0) red[wid] = s2;
  __syncthreads();
  float tot2 = 0.f;
#pragma unroll
  for (int i = 0; i < 8; i++) tot2 += red[i];
  float rs = rsqrtf(tot2 * (1.0f / HID) + 1e-5f);
  float* xo = xn + (size_t)row * HID;
  xo[t] = d0 * rs * gamma[t] + beta[t];
  xo[t + 256] = d1 * rs * gamma[t + 256] + beta[t + 256];
}

// ---------------- cumlogsumexp scan over sequence, cols 320..383 ----------------
__global__ void scan_kernel(const float* __restrict__ xn,
                            unsigned short* __restrict__ xt) {
  int b = blockIdx.x;
  int c = threadIdx.x;  // 0..63
  float m = -INFINITY, s = 0.f;
  for (int l = 0; l < SEQN; ++l) {
    size_t idx = ((size_t)(b * SEQN + l)) * HID + 320 + c;
    float v = 5.0f * xn[idx];
    float mn = fmaxf(m, v);
    s = s * __expf(m - mn) + __expf(v - mn);
    m = mn;
    xt[idx] = f32_to_bf16u((m + __logf(s)) * 0.2f);
  }
}

// ---------------- assemble xt (ident + shifted cols) as bf16 ----------------
__global__ __launch_bounds__(256) void xt_kernel(const float* __restrict__ xn,
                                                 unsigned short* __restrict__ xt) {
  int row = blockIdx.x;
  int lseq = row & (SEQN - 1);
  int t = threadIdx.x;
#pragma unroll
  for (int u = 0; u < 2; ++u) {
    int col = t + u * 256;
    if (col >= 320 && col < 384) continue;  // scan kernel owns these
    float v;
    if (col < 320)      v = xn[(size_t)row * HID + col];
    else if (col < 448) v = (lseq >= 1) ? xn[(size_t)(row - 1) * HID + col] : 0.f;
    else                v = (lseq >= 2) ? xn[(size_t)(row - 2) * HID + col] : 0.f;
    xt[(size_t)row * HID + col] = f32_to_bf16u(v);
  }
}

// -------- transpose + f32->bf16 weight conversion: src KxN -> dst NxK --------
__global__ __launch_bounds__(256) void cvtT_kernel(const float* __restrict__ src,
                                                   unsigned short* __restrict__ dst,
                                                   int K, int N) {
  int idx = blockIdx.x * 256 + threadIdx.x;
  if (idx < K * N) {
    int k = idx / N, n = idx % N;
    dst[(size_t)n * K + k] = f32_to_bf16u(src[idx]);
  }
}

// ---------------- bf16 WMMA GEMM (C^T orientation), K-chunk 32 ----------------
// Double-buffered LDS tiles filled by async global->LDS DMA.
// mode 0: GEMM1 epilogue (q,k bf16; v transposed; gelu(p) bf16)
// mode 1: GEMM2 epilogue (bias + f32 out)
__global__ __launch_bounds__(256) void gemm_kernel(
    const unsigned short* __restrict__ Xact, const unsigned short* __restrict__ Wt,
    int K, int N, int mode,
    unsigned short* __restrict__ qbuf, unsigned short* __restrict__ kbuf,
    unsigned short* __restrict__ vT, unsigned short* __restrict__ a2,
    float* __restrict__ outf, const float* __restrict__ bias) {
  __shared__ unsigned short lW[2][128 * 32];
  __shared__ unsigned short lX[2][128 * 32];
  int n0 = blockIdx.x * 128;
  int m0 = blockIdx.y * 128;
  int t = threadIdx.x;
  int wid = t >> 5, lane = t & 31;
  int l = lane & 15, h = lane >> 4;

  v8f acc[8];
#pragma unroll
  for (int n = 0; n < 8; n++)
#pragma unroll
    for (int j = 0; j < 8; j++) acc[n][j] = 0.f;

  int r = t >> 1, sg = (t & 1) * 16;
  const unsigned short* wrow = Wt + (size_t)(n0 + r) * K + sg;
  const unsigned short* xrow = Xact + (size_t)(m0 + r) * K + sg;
  unsigned lw0 = lds_addr32(&lW[0][r * 32 + sg]);
  unsigned lw1 = lds_addr32(&lW[1][r * 32 + sg]);
  unsigned lx0 = lds_addr32(&lX[0][r * 32 + sg]);
  unsigned lx1 = lds_addr32(&lX[1][r * 32 + sg]);

  // prologue: stage chunk 0 into buffer 0
  async_b128(lw0, wrow);
  async_b128(lw0 + 16, wrow + 8);
  async_b128(lx0, xrow);
  async_b128(lx0 + 16, xrow + 8);
  wait_async0();
  __syncthreads();

  int cur = 0;
  for (int k0 = 0; k0 < K; k0 += 32) {
    if (k0 + 32 < K) {  // stage next chunk into the other buffer (overlaps WMMAs)
      unsigned lw = cur ? lw0 : lw1;
      unsigned lx = cur ? lx0 : lx1;
      async_b128(lw, wrow + k0 + 32);
      async_b128(lw + 16, wrow + k0 + 40);
      async_b128(lx, xrow + k0 + 32);
      async_b128(lx + 16, xrow + k0 + 40);
    }
    v16bf af = load_afrag(&lW[cur][(wid * 16 + l) * 32], h);
#pragma unroll
    for (int mt = 0; mt < 8; mt++) {
      v16bf bf = load_bfrag(&lX[cur][(mt * 16 + l) * 32], h);
      acc[mt] = __builtin_amdgcn_wmma_f32_16x16x32_bf16(false, af, false, bf,
                                                        (short)0, acc[mt], false, false);
    }
    wait_async0();
    __syncthreads();
    cur ^= 1;
  }

#pragma unroll
  for (int mt = 0; mt < 8; mt++) {
#pragma unroll
    for (int rr = 0; rr < 8; rr++) {
      int n = n0 + wid * 16 + rr + 8 * h;  // output column
      int m = m0 + mt * 16 + l;            // output row
      float v = acc[mt][rr];
      if (mode == 0) {
        if (n < HID) {
          qbuf[(size_t)m * HID + n] = f32_to_bf16u(v);
        } else if (n < 2 * HID) {
          kbuf[(size_t)m * HID + (n - HID)] = f32_to_bf16u(v);
        } else if (n < 3 * HID) {
          int c = n - 2 * HID;
          int head = c >> 6, d = c & 63;
          int b = m >> 11, j = m & (SEQN - 1);
          vT[(((size_t)b * HEADSN + head) * HDIMN + d) * SEQN + j] = f32_to_bf16u(v);
        } else {
          float g = 0.5f * v * (1.0f + erff(v * 0.70710678118654752f));
          a2[(size_t)m * A2C + HID + (n - 3 * HID)] = f32_to_bf16u(g);
        }
      } else {
        outf[(size_t)m * HID + n] = v + bias[n];
      }
    }
  }
}

// ---------------- flash attention w/ causal + ALiBi, bf16 WMMA ----------------
__global__ __launch_bounds__(256) void attn_kernel(const unsigned short* __restrict__ qbuf,
                                                   const unsigned short* __restrict__ kbuf,
                                                   const unsigned short* __restrict__ vT,
                                                   unsigned short* __restrict__ a2) {
  __shared__ unsigned short lK[2][128 * 64];   // [j][d], 2x16 KB
  __shared__ unsigned short lVt[2][64 * 128];  // [d][j], 2x16 KB
  __shared__ unsigned short lP[8 * 16 * 128];  // per-wave P staging, 32 KB
  int ib = blockIdx.x;
  int bh = blockIdx.y;
  int b = bh >> 3, head = bh & 7;
  int t = threadIdx.x;
  int wid = t >> 5, lane = t & 31;
  int l = lane & 15, h = lane >> 4;
  int i0 = ib * 128;
  int r0 = i0 + wid * 16;

  // per-thread staging slots
  int rk = t >> 1, sgk = (t & 1) * 32;
  const unsigned short* krow = kbuf + ((size_t)(b * SEQN + rk)) * HID + head * HDIMN + sgk;
  int dv = t >> 2, sgv = (t & 3) * 32;
  const unsigned short* vrow = vT + (((size_t)b * HEADSN + head) * HDIMN + dv) * SEQN + sgv;
  unsigned lk0 = lds_addr32(&lK[0][rk * 64 + sgk]);
  unsigned lk1 = lds_addr32(&lK[1][rk * 64 + sgk]);
  unsigned lv0 = lds_addr32(&lVt[0][dv * 128 + sgv]);
  unsigned lv1 = lds_addr32(&lVt[1][dv * 128 + sgv]);

  // Q fragments (two 32-wide K-chunks of head_dim=64) straight from global
  const unsigned short* qbase = qbuf + ((size_t)(b * SEQN + r0 + l)) * HID + head * HDIMN;
  v16bf qf0 = load_afrag(qbase, h);
  v16bf qf1 = load_afrag(qbase + 32, h);

  v8f accO[4];
#pragma unroll
  for (int vt = 0; vt < 4; vt++)
#pragma unroll
    for (int j = 0; j < 8; j++) accO[vt][j] = 0.f;
  float m_run[8], l_run[8];
#pragma unroll
  for (int rr = 0; rr < 8; rr++) { m_run[rr] = -1e30f; l_run[rr] = 0.f; }
  float slope = exp2f(-(float)(head + 1));  // ALiBi slopes for 8 heads

  // prologue: stage j-block 0 into buffer 0 (K rows stride HID; Vt rows stride SEQN)
#pragma unroll
  for (int q4 = 0; q4 < 4; q4++) {
    async_b128(lk0 + q4 * 16, krow + q4 * 8);
    async_b128(lv0 + q4 * 16, vrow + q4 * 8);
  }
  wait_async0();
  __syncthreads();

  int cur = 0;
  for (int jb = 0; jb <= ib; ++jb) {
    int j0 = jb * 128;
    if (jb < ib) {  // stage next j-block into the other buffer (overlaps compute)
      unsigned lk = cur ? lk0 : lk1;
      unsigned lv = cur ? lv0 : lv1;
      const unsigned short* kn = krow + (size_t)(j0 + 128) * HID;
      const unsigned short* vn = vrow + (j0 + 128);
#pragma unroll
      for (int q4 = 0; q4 < 4; q4++) {
        async_b128(lk + q4 * 16, kn + q4 * 8);
        async_b128(lv + q4 * 16, vn + q4 * 8);
      }
    }

    // S = Q * K^T for this wave's 16x128 strip
    v8f S[8];
#pragma unroll
    for (int n = 0; n < 8; n++) {
#pragma unroll
      for (int j = 0; j < 8; j++) S[n][j] = 0.f;
      v16bf b0 = load_bfrag(&lK[cur][(n * 16 + l) * 64], h);
      v16bf b1 = load_bfrag(&lK[cur][(n * 16 + l) * 64 + 32], h);
      S[n] = __builtin_amdgcn_wmma_f32_16x16x32_bf16(false, qf0, false, b0,
                                                     (short)0, S[n], false, false);
      S[n] = __builtin_amdgcn_wmma_f32_16x16x32_bf16(false, qf1, false, b1,
                                                     (short)0, S[n], false, false);
    }

    bool diag = (jb == ib);
#pragma unroll
    for (int n = 0; n < 8; n++) {
#pragma unroll
      for (int rr = 0; rr < 8; rr++) {
        int row = r0 + rr + 8 * h;
        int col = j0 + n * 16 + l;
        float v = S[n][rr] * 0.125f + slope * (float)col;
        if (diag && col > row) v = -1e30f;
        S[n][rr] = v;
      }
    }

    // online softmax (rows live in fixed lane half; reduce within 16 lanes)
    unsigned short* myP = lP + wid * (16 * 128);
#pragma unroll
    for (int rr = 0; rr < 8; rr++) {
      float mx = S[0][rr];
#pragma unroll
      for (int n = 1; n < 8; n++) mx = fmaxf(mx, S[n][rr]);
#pragma unroll
      for (int msk = 1; msk < 16; msk <<= 1) mx = fmaxf(mx, __shfl_xor(mx, msk));
      float mnew = fmaxf(m_run[rr], mx);
      float srow = 0.f;
#pragma unroll
      for (int n = 0; n < 8; n++) {
        float e = __expf(S[n][rr] - mnew);
        S[n][rr] = e;
        srow += e;
      }
#pragma unroll
      for (int msk = 1; msk < 16; msk <<= 1) srow += __shfl_xor(srow, msk);
      float fac = __expf(m_run[rr] - mnew);
      l_run[rr] = l_run[rr] * fac + srow;
      m_run[rr] = mnew;
#pragma unroll
      for (int vt = 0; vt < 4; vt++) accO[vt][rr] = accO[vt][rr] * fac;
    }

    // stage P (C-layout -> row-major [m][j]); per-wave private, LDS is in-order
    // within a wave so no workgroup barrier is needed before the reload.
#pragma unroll
    for (int n = 0; n < 8; n++)
#pragma unroll
      for (int rr = 0; rr < 8; rr++)
        myP[(rr + 8 * h) * 128 + n * 16 + l] = f32_to_bf16u(S[n][rr]);

    // O += P * V  (P: A-frags from myP; V: B-frags from lVt, both b128)
#pragma unroll
    for (int c = 0; c < 4; c++) {
      v16bf pf = load_afrag(&myP[l * 128 + 32 * c], h);
#pragma unroll
      for (int vt = 0; vt < 4; vt++) {
        v16bf vf = load_bfrag(&lVt[cur][(vt * 16 + l) * 128 + 32 * c], h);
        accO[vt] = __builtin_amdgcn_wmma_f32_16x16x32_bf16(false, pf, false, vf,
                                                           (short)0, accO[vt], false, false);
      }
    }
    wait_async0();
    __syncthreads();
    cur ^= 1;
  }

  // normalize + write O into a2[:, 0:512]
#pragma unroll
  for (int vt = 0; vt < 4; vt++) {
#pragma unroll
    for (int rr = 0; rr < 8; rr++) {
      int row = b * SEQN + r0 + rr + 8 * h;
      int col = head * HDIMN + vt * 16 + l;
      float o = accO[vt][rr] / l_run[rr];
      a2[(size_t)row * A2C + col] = f32_to_bf16u(o);
    }
  }
}

extern "C" void kernel_launch(void* const* d_in, const int* in_sizes, int n_in,
                              void* d_out, int out_size, void* d_ws, size_t ws_size,
                              hipStream_t stream) {
  (void)in_sizes; (void)n_in; (void)out_size; (void)ws_size;
  const float* x     = (const float*)d_in[0];
  const float* gamma = (const float*)d_in[1];
  const float* beta  = (const float*)d_in[2];
  const float* w_in  = (const float*)d_in[3];
  const float* w_out = (const float*)d_in[4];
  const float* b_out = (const float*)d_in[5];
  float* out = (float*)d_out;

  char* ws = (char*)d_ws;
  float*          xn     = (float*)ws;                          //  8,388,608 B
  unsigned short* xt     = (unsigned short*)(ws + 8388608);     //  4,194,304 B
  unsigned short* wT_in  = (unsigned short*)(ws + 12582912);    //  3,670,016 B (3584x512)
  unsigned short* wT_out = (unsigned short*)(ws + 16252928);    //  2,621,440 B (512x2560)
  unsigned short* qbuf   = (unsigned short*)(ws + 18874368);    //  4,194,304 B
  unsigned short* kbuf   = (unsigned short*)(ws + 23068672);    //  4,194,304 B
  unsigned short* vT     = (unsigned short*)(ws + 27262976);    //  4,194,304 B
  unsigned short* a2     = (unsigned short*)(ws + 31457280);    // 20,971,520 B -> 52,428,800 total

  ln_kernel<<<ROWSN, 256, 0, stream>>>(x, gamma, beta, xn);
  scan_kernel<<<BATCHN, 64, 0, stream>>>(xn, xt);
  xt_kernel<<<ROWSN, 256, 0, stream>>>(xn, xt);
  cvtT_kernel<<<(HID * QKVN + 255) / 256, 256, 0, stream>>>(w_in, wT_in, HID, QKVN);
  cvtT_kernel<<<(A2C * HID + 255) / 256, 256, 0, stream>>>(w_out, wT_out, A2C, HID);
  gemm_kernel<<<dim3(QKVN / 128, ROWSN / 128), 256, 0, stream>>>(
      xt, wT_in, HID, QKVN, 0, qbuf, kbuf, vT, a2, nullptr, nullptr);
  attn_kernel<<<dim3(SEQN / 128, BATCHN * HEADSN), 256, 0, stream>>>(qbuf, kbuf, vT, a2);
  gemm_kernel<<<dim3(HID / 128, ROWSN / 128), 256, 0, stream>>>(
      a2, wT_out, A2C, HID, 1, nullptr, nullptr, nullptr, nullptr, out, b_out);
}
